// MultiHeadAttention_29205777612996
// MI455X (gfx1250) — compile-verified, matrix operand order checked by reading, not run on hardware
//
#include <hip/hip_runtime.h>
#include <hip/hip_bf16.h>
#include <math.h>

#define BATCH 4
#define NCTX  2048
#define DIM   1024
#define HEADS 16
#define HD    64
#define MTOT  (BATCH * NCTX)   /* 8192 */
#define NQKV  (3 * DIM)        /* 3072 */
#define KDIM  DIM
#define TS64  (KDIM / 64)      /* 16 staged K-tiles */

typedef __attribute__((ext_vector_type(16))) __bf16 v16bf;
typedef __attribute__((ext_vector_type(8)))  float  v8f;

union FragU { v16bf v; uint4 q[2]; };

#if __has_builtin(__builtin_amdgcn_global_load_async_to_lds_b128) && \
    __has_builtin(__builtin_amdgcn_s_wait_asynccnt)
#define HAVE_ASYNC_LDS 1
typedef int v4i __attribute__((vector_size(16)));
typedef __attribute__((address_space(1))) v4i* gptr_v4i;
typedef __attribute__((address_space(3))) v4i* lptr_v4i;
#else
#define HAVE_ASYNC_LDS 0
#endif

__device__ __forceinline__ unsigned short f2bf(float f) {
  unsigned int u = __float_as_uint(f);
  u += 0x7FFFu + ((u >> 16) & 1u);   // round-to-nearest-even
  return (unsigned short)(u >> 16);
}

// Load a 16-bit A/B WMMA fragment: two contiguous 8-element (16B) runs.
// CDNA5 16-bit layout: element e of lane l -> K = (e/8)*16 + (l/16)*8 + e%8.
__device__ __forceinline__ v16bf load_frag(const unsigned short* lo,
                                           const unsigned short* hi) {
  FragU f;
  f.q[0] = *(const uint4*)lo;
  f.q[1] = *(const uint4*)hi;
  return f.v;
}

__device__ __forceinline__ v8f wmma_bf16(v16bf a, v16bf b, v8f c) {
  return __builtin_amdgcn_wmma_f32_16x16x32_bf16(
      /*neg_a=*/false, a, /*neg_b=*/false, b,
      /*c_mod=*/(short)0, c, /*reuse_a=*/false, /*reuse_b=*/false);
}

// 16-byte global -> LDS copy. Async path on gfx1250 (ASYNCcnt-tracked),
// falls back to load+ds_store if the builtin is unavailable.
__device__ __forceinline__ void async_copy16(unsigned short* lds_dst,
                                             const unsigned short* g_src) {
#if HAVE_ASYNC_LDS
  __builtin_amdgcn_global_load_async_to_lds_b128(
      (gptr_v4i)(unsigned long long)g_src,
      (lptr_v4i)(unsigned int)(unsigned long long)lds_dst,
      0, 0);
#else
  *(uint4*)lds_dst = *(const uint4*)g_src;
#endif
}

__device__ __forceinline__ void wait_async0() {
#if HAVE_ASYNC_LDS
  __builtin_amdgcn_s_wait_asynccnt(0);
#endif
}

// ---------------------------------------------------------------------------
// Kernel 0a: fp32 -> bf16 pack (2 elements / thread, packed dword store)
// ---------------------------------------------------------------------------
__global__ void pack_bf16_kernel(const float* __restrict__ src,
                                 unsigned int* __restrict__ dst, int n2) {
  int i = blockIdx.x * blockDim.x + threadIdx.x;
  if (i < n2) {
    float2 f = ((const float2*)src)[i];
    dst[i] = (unsigned int)f2bf(f.x) | ((unsigned int)f2bf(f.y) << 16);
  }
}

// ---------------------------------------------------------------------------
// Kernel 0b: fp32 [rows,cols] -> bf16 transposed [cols,rows]
// ---------------------------------------------------------------------------
__global__ void transpose_bf16_kernel(const float* __restrict__ src,
                                      unsigned short* __restrict__ dst,
                                      int rows, int cols) {
  int i = blockIdx.x * blockDim.x + threadIdx.x;
  if (i < rows * cols) {
    int c = i / rows;   // output row  (original column)
    int r = i % rows;   // output col  (original row)
    dst[(size_t)c * rows + r] = f2bf(src[(size_t)r * cols + c]);
  }
}

// ---------------------------------------------------------------------------
// Shared GEMM core: 4-wave block, block tile 128M x 64N, K=1024.
// B staged in LDS as 64N x 64K bf16 tiles (8KB) via async copies, double
// buffered with static buffer pointers (2x unrolled). Each wave computes a
// 32M x 64N sub-tile: 16 WMMA per staged tile between wait/barrier pairs.
// ---------------------------------------------------------------------------
#define BT64 (64 * 64)

__device__ __forceinline__ void gemm_copy_btile(const unsigned short* __restrict__ wT,
                                                int nbase, int kk64,
                                                unsigned short* bufp) {
  const int tid = threadIdx.x;
  #pragma unroll
  for (int i = 0; i < 4; i++) {
    int chunk = tid + i * 128;           // 512 x 16B chunks = 8KB
    int row = chunk >> 3;                // 0..63  (N within tile)
    int c = chunk & 7;                   // 0..7   (8-short K group)
    async_copy16(bufp + row * 64 + c * 8,
                 wT + (size_t)(nbase + row) * KDIM + kk64 + c * 8);
  }
}

// One 32-deep K step: A frags from global, B frags from LDS tile (koff 0/32).
__device__ __forceinline__ void gemm_step(const unsigned short* bb,
                                          const unsigned short* arow0,
                                          const unsigned short* arow1,
                                          int kk, int koff, int lh, int lm,
                                          v8f acc[2][4]) {
  v16bf a0 = load_frag(arow0 + kk + lh * 8, arow0 + kk + 16 + lh * 8);
  v16bf a1 = load_frag(arow1 + kk + lh * 8, arow1 + kk + 16 + lh * 8);
  #pragma unroll
  for (int ni = 0; ni < 4; ni++) {
    const unsigned short* br = bb + (ni * 16 + lm) * 64 + koff;
    v16bf bf = load_frag(br + lh * 8, br + 16 + lh * 8);
    acc[0][ni] = wmma_bf16(a0, bf, acc[0][ni]);
    acc[1][ni] = wmma_bf16(a1, bf, acc[1][ni]);
  }
}

__device__ __forceinline__ void gemm_mainloop(const unsigned short* __restrict__ abase,
                                              const unsigned short* __restrict__ wT,
                                              int rowblock, int nbase,
                                              unsigned short* Bbuf,  // shared, 2*BT64
                                              v8f acc[2][4]) {
  const int tid = threadIdx.x;
  const int w = tid >> 5;
  const int lane = tid & 31;
  const int lh = lane >> 4, lm = lane & 15;
  const int rowbase = rowblock + w * 32;

  v8f z = {};
  #pragma unroll
  for (int i = 0; i < 2; i++)
    #pragma unroll
    for (int j = 0; j < 4; j++) acc[i][j] = z;

  const unsigned short* arow0 = abase + (size_t)(rowbase + lm) * KDIM;
  const unsigned short* arow1 = abase + (size_t)(rowbase + 16 + lm) * KDIM;

  gemm_copy_btile(wT, nbase, 0, Bbuf);
  wait_async0();
  __syncthreads();

  for (int t = 0; t < TS64; t += 2) {
    // refill back buffer while front buffer feeds 16 WMMAs
    gemm_copy_btile(wT, nbase, (t + 1) * 64, Bbuf + BT64);
    gemm_step(Bbuf, arow0, arow1, t * 64, 0, lh, lm, acc);
    gemm_step(Bbuf, arow0, arow1, t * 64 + 32, 32, lh, lm, acc);
    wait_async0();
    __syncthreads();

    if (t + 2 < TS64)
      gemm_copy_btile(wT, nbase, (t + 2) * 64, Bbuf);
    gemm_step(Bbuf + BT64, arow0, arow1, (t + 1) * 64, 0, lh, lm, acc);
    gemm_step(Bbuf + BT64, arow0, arow1, (t + 1) * 64 + 32, 32, lh, lm, acc);
    wait_async0();
    __syncthreads();
  }
}

// ---------------------------------------------------------------------------
// Kernel 1: QKV projection GEMM (M=8192, N=3072, K=1024), bf16 WMMA.
// Scatters into Q (pre-scaled by 1/sqrt(d)), K, and V^T.
// ---------------------------------------------------------------------------
__global__ __launch_bounds__(128) void qkv_gemm_kernel(
    const unsigned short* __restrict__ xb,    // [8192,1024] bf16
    const unsigned short* __restrict__ wT,    // [3072,1024] bf16 (W^T)
    unsigned short* __restrict__ qd,          // [64,2048,64] bf16, pre-scaled
    unsigned short* __restrict__ kd,          // [64,2048,64] bf16
    unsigned short* __restrict__ vt)          // [64,64,2048] bf16 (V^T)
{
  __shared__ alignas(16) unsigned short Bbuf[2 * BT64];
  const int lane = threadIdx.x & 31;
  const int lh = lane >> 4, lm = lane & 15;
  const int nbase = blockIdx.x * 64;
  const int rowblock = blockIdx.y * 128;
  const int rowbase = rowblock + (threadIdx.x >> 5) * 32;

  v8f acc[2][4];
  gemm_mainloop(xb, wT, rowblock, nbase, Bbuf, acc);

  // Tile is 64 cols wide, 64-aligned -> one head, one of Q/K/V.
  const int section = nbase >> 10;          // 0=Q, 1=K, 2=V
  const int h = (nbase & 1023) >> 6;
  #pragma unroll
  for (int mi = 0; mi < 2; mi++)
    #pragma unroll
    for (int ni = 0; ni < 4; ni++)
      #pragma unroll
      for (int r = 0; r < 8; r++) {
        float v = acc[mi][ni][r];
        int token = rowbase + mi * 16 + r + 8 * lh;     // C layout: M=r+8*lh
        int b = token >> 11, n = token & (NCTX - 1);
        int d = ni * 16 + lm;                           // C layout: N=lm
        size_t bh = (size_t)(b * HEADS + h);
        if (section == 0)
          qd[(bh * NCTX + n) * HD + d] = f2bf(v * 0.125f);  // fold 1/sqrt(64)
        else if (section == 1)
          kd[(bh * NCTX + n) * HD + d] = f2bf(v);
        else
          vt[(bh * HD + d) * NCTX + n] = f2bf(v);
      }
}

// ---------------------------------------------------------------------------
// Kernel 2: flash attention. One wave = 16 query rows of one (b,h).
// Streams 32 keys/values per iter: 4 score WMMAs, online softmax
// (shfl_xor row reductions within 16-lane halves), P via LDS -> 4 PV WMMAs.
// ---------------------------------------------------------------------------
__global__ __launch_bounds__(32) void attn_kernel(
    const unsigned short* __restrict__ qd,
    const unsigned short* __restrict__ kd,
    const unsigned short* __restrict__ vt,
    unsigned short* __restrict__ oattn)      // [8192,1024] bf16
{
  __shared__ alignas(16) unsigned short Pbuf[16 * 32];

  const int lane = threadIdx.x;
  const int lh = lane >> 4, lm = lane & 15;
  const int bh = blockIdx.x;
  const int rowbase = blockIdx.y * 16;

  const unsigned short* qb = qd + (size_t)bh * NCTX * HD;
  const unsigned short* kbp = kd + (size_t)bh * NCTX * HD;
  const unsigned short* vbp = vt + (size_t)bh * HD * NCTX;

  const unsigned short* qrow = qb + (size_t)(rowbase + lm) * HD;
  v16bf qa0 = load_frag(qrow + lh * 8, qrow + 16 + lh * 8);        // d 0..31
  v16bf qa1 = load_frag(qrow + 32 + lh * 8, qrow + 48 + lh * 8);   // d 32..63

  v8f o0 = {}, o1 = {}, o2 = {}, o3 = {};
  float m_i[8], l_i[8];
  #pragma unroll
  for (int r = 0; r < 8; r++) { m_i[r] = -1e30f; l_i[r] = 0.0f; }

  for (int jb = 0; jb < NCTX; jb += 32) {
    // ---- scores S[16 x 32] = Qs @ K^T (two 16-col C tiles) ----
    const unsigned short* krow0 = kbp + (size_t)(jb + lm) * HD;
    const unsigned short* krow1 = kbp + (size_t)(jb + 16 + lm) * HD;
    v16bf k00 = load_frag(krow0 + lh * 8, krow0 + 16 + lh * 8);
    v16bf k01 = load_frag(krow0 + 32 + lh * 8, krow0 + 48 + lh * 8);
    v16bf k10 = load_frag(krow1 + lh * 8, krow1 + 16 + lh * 8);
    v16bf k11 = load_frag(krow1 + 32 + lh * 8, krow1 + 48 + lh * 8);

    v8f s0 = {}, s1 = {};
    s0 = wmma_bf16(qa0, k00, s0);
    s0 = wmma_bf16(qa1, k01, s0);
    s1 = wmma_bf16(qa0, k10, s1);
    s1 = wmma_bf16(qa1, k11, s1);

    // ---- online softmax (rows align with C layout: M = r + 8*lh) ----
    float alpha[8];
    #pragma unroll
    for (int r = 0; r < 8; r++) {
      float mx = fmaxf(s0[r], s1[r]);
      mx = fmaxf(mx, __shfl_xor(mx, 1));
      mx = fmaxf(mx, __shfl_xor(mx, 2));
      mx = fmaxf(mx, __shfl_xor(mx, 4));
      mx = fmaxf(mx, __shfl_xor(mx, 8));
      float mnew = fmaxf(m_i[r], mx);
      float a = __expf(m_i[r] - mnew);
      float p0 = __expf(s0[r] - mnew);
      float p1 = __expf(s1[r] - mnew);
      float rs = p0 + p1;
      rs += __shfl_xor(rs, 1);
      rs += __shfl_xor(rs, 2);
      rs += __shfl_xor(rs, 4);
      rs += __shfl_xor(rs, 8);
      l_i[r] = l_i[r] * a + rs;
      m_i[r] = mnew;
      alpha[r] = a;
      // C layout -> LDS (row-major 16x32 bf16); DS ops in-order per wave.
      Pbuf[(r + 8 * lh) * 32 + lm] = f2bf(p0);
      Pbuf[(r + 8 * lh) * 32 + 16 + lm] = f2bf(p1);
    }
    #pragma unroll
    for (int r = 0; r < 8; r++) {
      o0[r] *= alpha[r]; o1[r] *= alpha[r];
      o2[r] *= alpha[r]; o3[r] *= alpha[r];
    }

    // P back in A-fragment layout (row = lm, contiguous K runs).
    v16bf pA = load_frag(&Pbuf[lm * 32 + lh * 8], &Pbuf[lm * 32 + 16 + lh * 8]);

    // ---- O += P @ V : B fragments from V^T rows (contiguous in j) ----
    const unsigned short* v0r = vbp + (size_t)(0 + lm) * NCTX + jb;
    const unsigned short* v1r = vbp + (size_t)(16 + lm) * NCTX + jb;
    const unsigned short* v2r = vbp + (size_t)(32 + lm) * NCTX + jb;
    const unsigned short* v3r = vbp + (size_t)(48 + lm) * NCTX + jb;
    v16bf vf0 = load_frag(v0r + lh * 8, v0r + 16 + lh * 8);
    v16bf vf1 = load_frag(v1r + lh * 8, v1r + 16 + lh * 8);
    v16bf vf2 = load_frag(v2r + lh * 8, v2r + 16 + lh * 8);
    v16bf vf3 = load_frag(v3r + lh * 8, v3r + 16 + lh * 8);

    o0 = wmma_bf16(pA, vf0, o0);
    o1 = wmma_bf16(pA, vf1, o1);
    o2 = wmma_bf16(pA, vf2, o2);
    o3 = wmma_bf16(pA, vf3, o3);
  }

  // ---- epilogue: O / l_i -> oattn [token, h*64 + d] bf16 ----
  const int b = bh >> 4, h = bh & 15;
  #pragma unroll
  for (int r = 0; r < 8; r++) {
    float inv = 1.0f / l_i[r];
    int n = rowbase + r + 8 * lh;
    size_t base = ((size_t)b * NCTX + n) * DIM + h * HD;
    oattn[base + 0 * 16 + lm] = f2bf(o0[r] * inv);
    oattn[base + 1 * 16 + lm] = f2bf(o1[r] * inv);
    oattn[base + 2 * 16 + lm] = f2bf(o2[r] * inv);
    oattn[base + 3 * 16 + lm] = f2bf(o3[r] * inv);
  }
}

// ---------------------------------------------------------------------------
// Kernel 3: output projection (M=8192, N=1024, K=1024) + bias, fp32 out.
// ---------------------------------------------------------------------------
__global__ __launch_bounds__(128) void out_gemm_kernel(
    const unsigned short* __restrict__ ob,    // [8192,1024] bf16
    const unsigned short* __restrict__ wT,    // [1024,1024] bf16 (W^T)
    const float* __restrict__ bias,           // [1024]
    float* __restrict__ out)                  // [8192,1024] fp32
{
  __shared__ alignas(16) unsigned short Bbuf[2 * BT64];
  const int lane = threadIdx.x & 31;
  const int lh = lane >> 4, lm = lane & 15;
  const int nbase = blockIdx.x * 64;
  const int rowblock = blockIdx.y * 128;
  const int rowbase = rowblock + (threadIdx.x >> 5) * 32;

  v8f acc[2][4];
  gemm_mainloop(ob, wT, rowblock, nbase, Bbuf, acc);

  #pragma unroll
  for (int mi = 0; mi < 2; mi++)
    #pragma unroll
    for (int ni = 0; ni < 4; ni++) {
      int Nc = nbase + ni * 16 + lm;
      float bv = bias[Nc];
      #pragma unroll
      for (int r = 0; r < 8; r++) {
        int M = rowbase + mi * 16 + r + 8 * lh;
        out[(size_t)M * DIM + Nc] = acc[mi][ni][r] + bv;
      }
    }
}

// ---------------------------------------------------------------------------
extern "C" void kernel_launch(void* const* d_in, const int* in_sizes, int n_in,
                              void* d_out, int out_size, void* d_ws,
                              size_t ws_size, hipStream_t stream) {
  (void)in_sizes; (void)n_in; (void)out_size; (void)ws_size;
  const float* x     = (const float*)d_in[0];
  const float* w_in  = (const float*)d_in[1];
  const float* w_out = (const float*)d_in[2];
  const float* b_out = (const float*)d_in[3];
  float* out = (float*)d_out;

  char* ws = (char*)d_ws;
  size_t off = 0;
  unsigned short* xb    = (unsigned short*)(ws + off); off += (size_t)MTOT * DIM * 2;
  unsigned short* w_inT = (unsigned short*)(ws + off); off += (size_t)NQKV * DIM * 2;
  unsigned short* w_oT  = (unsigned short*)(ws + off); off += (size_t)DIM * DIM * 2;
  unsigned short* qd    = (unsigned short*)(ws + off); off += (size_t)MTOT * DIM * 2;
  unsigned short* kd    = (unsigned short*)(ws + off); off += (size_t)MTOT * DIM * 2;
  unsigned short* vt    = (unsigned short*)(ws + off); off += (size_t)MTOT * DIM * 2;
  unsigned short* oat   = (unsigned short*)(ws + off); off += (size_t)MTOT * DIM * 2;

  // 0) precision conversion / layout transforms
  {
    int n2 = MTOT * DIM / 2;
    pack_bf16_kernel<<<(n2 + 255) / 256, 256, 0, stream>>>(x, (unsigned int*)xb, n2);
  }
  transpose_bf16_kernel<<<(DIM * NQKV + 255) / 256, 256, 0, stream>>>(w_in, w_inT, DIM, NQKV);
  transpose_bf16_kernel<<<(DIM * DIM + 255) / 256, 256, 0, stream>>>(w_out, w_oT, DIM, DIM);

  // 1) QKV projection -> Q(scaled)/K/V^T   (async-LDS staged B, 4-wave blocks)
  qkv_gemm_kernel<<<dim3(NQKV / 64, MTOT / 128), 128, 0, stream>>>(xb, w_inT, qd, kd, vt);

  // 2) flash attention per (b,h) x 16-row block
  attn_kernel<<<dim3(BATCH * HEADS, NCTX / 16), 32, 0, stream>>>(qd, kd, vt, oat);

  // 3) output projection + bias
  out_gemm_kernel<<<dim3(DIM / 64, MTOT / 128), 128, 0, stream>>>(oat, w_oT, b_out, out);
}
